// GATWithDropout_8177617732165
// MI455X (gfx1250) — matrix operand order repaired and checked by reading, not compiled, not run on hardware
//
#include <hip/hip_runtime.h>
#include <hip/hip_bf16.h>

// ---------------- problem constants (match reference) ----------------
#define F_IN     128
#define HID      32
#define HEADS    4
#define OUT_DIM  10
#define NGRAPH   64
#define NEG_SLOPE 0.2f

// ---------------- WMMA vector types (gfx1250, wave32) ----------------
typedef __attribute__((ext_vector_type(16))) __bf16 v16bf;
typedef __attribute__((ext_vector_type(8)))  __bf16 v8bf;
typedef __attribute__((ext_vector_type(8)))  float  v8f;

union BF16Frag { v16bf v; v8bf h[2]; };

static __device__ __forceinline__ void atomicMaxFloat(float* addr, float val) {
  // classic signed/unsigned trick: works for any float incl. -inf init
  if (val >= 0.0f) atomicMax((int*)addr, __float_as_int(val));
  else             atomicMin((unsigned int*)addr, __float_as_uint(val));
}

// =====================================================================
// GEMM: C[M][Nc] = A[M][K](bf16) x Bt[Nc][K](bf16)^T, f32 accumulate.
// One wave32 per 16x16 output tile, K swept in 32-element WMMA steps.
// Fragment layouts per CDNA5 ISA 7.12.2:
//   A (16x32 bf16): lane half h=l>>4, row=l&15; K runs [8h,8h+8) and [8h+16,8h+24)
//   B (32x16 bf16): col=l&15; K runs [8h,8h+8) and [8h+16,8h+24)
//   C (16x16 f32):  vgpr r -> row = 8h + r, col = l&15
// =====================================================================
__global__ __launch_bounds__(256) void wmma_gemm_bf16_kernel(
    const __bf16* __restrict__ A, const __bf16* __restrict__ Bt,
    float* __restrict__ C, int M, int K, int Nc)
{
  const int lane    = threadIdx.x & 31;
  const int wave    = (blockIdx.x * blockDim.x + threadIdx.x) >> 5;
  const int tiles_n = Nc >> 4;
  const int tile_m  = wave / tiles_n;
  const int tile_n  = wave - tile_m * tiles_n;
  if (tile_m * 16 >= M) return;           // wave-uniform: EXEC stays all-1s

  const int half = lane >> 4;
  const int l16  = lane & 15;
  const int row  = tile_m * 16 + l16;     // A row for this lane
  const int col  = tile_n * 16 + l16;     // B col / C col for this lane

  const __bf16* __restrict__ arow = A  + (size_t)row * K;
  const __bf16* __restrict__ brow = Bt + (size_t)col * K;
  const int kh = half * 8;

  v8f acc = {};
  for (int kb = 0; kb < K; kb += 32) {
    BF16Frag a, b;
    a.h[0] = *(const v8bf*)(arow + kb + kh);        // K = kh .. kh+7
    a.h[1] = *(const v8bf*)(arow + kb + kh + 16);   // K = kh+16 .. kh+23
    b.h[0] = *(const v8bf*)(brow + kb + kh);        // K = kh .. kh+7
    b.h[1] = *(const v8bf*)(brow + kb + kh + 16);   // K = kh+16 .. kh+23
    acc = __builtin_amdgcn_wmma_f32_16x16x32_bf16(
        /*neg_a=*/false, a.v, /*neg_b=*/false, b.v,
        /*c_mod=*/(short)0, acc, /*reuse_a=*/false, /*reuse_b=*/false);
  }

  const int r0 = tile_m * 16 + half * 8;
  #pragma unroll
  for (int r = 0; r < 8; ++r)
    C[(size_t)(r0 + r) * Nc + col] = acc[r];
}

// ---------------- prep / small kernels ----------------
__global__ void cast_bf16_kernel(const float* __restrict__ in,
                                 __bf16* __restrict__ out, int n) {
  int i = blockIdx.x * blockDim.x + threadIdx.x;
  if (i < n) out[i] = (__bf16)in[i];
}

// in [rows][cols] f32 -> out [cols][rows] bf16   (B-transpose for WMMA)
__global__ void transpose_cast_kernel(const float* __restrict__ in,
                                      __bf16* __restrict__ out,
                                      int rows, int cols) {
  int i = blockIdx.x * blockDim.x + threadIdx.x;
  if (i < rows * cols) {
    int r = i / cols, c = i - r * cols;
    out[(size_t)c * rows + r] = (__bf16)in[i];
  }
}

__global__ void fillf_kernel(float* __restrict__ p, float v, int n) {
  int i = blockIdx.x * blockDim.x + threadIdx.x;
  if (i < n) p[i] = v;
}

// per (node,head): alpha_src/dst dot products over C features
__global__ void alpha_kernel(const float* __restrict__ h,
                             const float* __restrict__ a_src,
                             const float* __restrict__ a_dst,
                             float* __restrict__ as, float* __restrict__ ad,
                             int n, int H, int C) {
  int i = blockIdx.x * blockDim.x + threadIdx.x;
  if (i >= n * H) return;
  int node = i / H, hd = i - node * H;
  const float* hp = h + (size_t)node * H * C + hd * C;
  const float* sp = a_src + hd * C;
  const float* dp = a_dst + hd * C;
  float s = 0.f, d = 0.f;
  for (int c = 0; c < C; ++c) { float v = hp[c]; s += v * sp[c]; d += v * dp[c]; }
  as[i] = s; ad[i] = d;
}

// pass 1: leaky-relu logits, segment max by dst (atomic)
__global__ void edge_logit_max_kernel(const int* __restrict__ esrc,
                                      const int* __restrict__ edst,
                                      const float* __restrict__ as,
                                      const float* __restrict__ ad,
                                      float* __restrict__ logits,
                                      float* __restrict__ m,
                                      int E, int Et, int H) {
  int i = blockIdx.x * blockDim.x + threadIdx.x;
  if (i >= Et * H) return;
  int e = i / H, hd = i - e * H;
  int s = (e < E) ? esrc[e] : (e - E);   // self-loops appended
  int d = (e < E) ? edst[e] : (e - E);
  float x = as[s * H + hd] + ad[d * H + hd];
  x = (x > 0.f) ? x : NEG_SLOPE * x;
  logits[i] = x;
  atomicMaxFloat(&m[d * H + hd], x);
}

// pass 2: e = exp(logit - m[d]) (overwrites logits), segment sum denom
__global__ void edge_expsum_kernel(const int* __restrict__ edst,
                                   float* __restrict__ logits,
                                   const float* __restrict__ m,
                                   float* __restrict__ denom,
                                   int E, int Et, int H) {
  int i = blockIdx.x * blockDim.x + threadIdx.x;
  if (i >= Et * H) return;
  int e = i / H, hd = i - e * H;
  int d = (e < E) ? edst[e] : (e - E);
  float ev = __expf(logits[i] - m[d * H + hd]);
  logits[i] = ev;
  atomicAdd(&denom[d * H + hd], ev);
}

// pass 3: out[d] += alpha * h[s], one wave per edge (L2-resident gathers)
__global__ void edge_scatter_kernel(const int* __restrict__ esrc,
                                    const int* __restrict__ edst,
                                    const float* __restrict__ eexp,
                                    const float* __restrict__ denom,
                                    const float* __restrict__ h,
                                    float* __restrict__ out,
                                    int E, int Et, int H, int C) {
  int lane = threadIdx.x & 31;
  int e = (blockIdx.x * blockDim.x + threadIdx.x) >> 5;
  if (e >= Et) return;
  int s = (e < E) ? esrc[e] : (e - E);
  int d = (e < E) ? edst[e] : (e - E);
  int F = H * C;
  for (int f = lane; f < F; f += 32) {
    int hd = f / C;
    float alpha = eexp[e * H + hd] / denom[d * H + hd];
    atomicAdd(&out[(size_t)d * F + f], alpha * h[(size_t)s * F + f]);
  }
}

// relu(out + bias) -> bf16 (input to next WMMA GEMM)
__global__ void relu_bias_cast_kernel(const float* __restrict__ in,
                                      const float* __restrict__ bias,
                                      __bf16* __restrict__ outb, int n, int F) {
  int i = blockIdx.x * blockDim.x + threadIdx.x;
  if (i >= n * F) return;
  float v = in[i] + bias[i % F];
  outb[i] = (__bf16)(v > 0.f ? v : 0.f);
}

// relu(out2 + b2) then atomic mean-pool accumulation by graph
__global__ void relu_pool_kernel(const float* __restrict__ in,
                                 const float* __restrict__ bias,
                                 const int* __restrict__ batch,
                                 float* __restrict__ pooled,
                                 float* __restrict__ cnt, int n, int C) {
  int i = blockIdx.x * blockDim.x + threadIdx.x;
  if (i >= n * C) return;
  int node = i / C, c = i - node * C;
  float v = in[i] + bias[c];
  v = v > 0.f ? v : 0.f;
  int g = batch[node];
  atomicAdd(&pooled[g * C + c], v);
  if (c == 0) atomicAdd(&cnt[g], 1.0f);
}

__global__ void fc_kernel(const float* __restrict__ pooled,
                          const float* __restrict__ cnt,
                          const float* __restrict__ fw,
                          const float* __restrict__ fb,
                          float* __restrict__ out) {
  int i = blockIdx.x * blockDim.x + threadIdx.x;
  if (i >= NGRAPH * OUT_DIM) return;
  int g = i / OUT_DIM, o = i - g * OUT_DIM;
  float inv = 1.0f / fmaxf(cnt[g], 1.0f);
  float acc = fb[o];
  for (int c = 0; c < HID; ++c)
    acc += pooled[g * HID + c] * inv * fw[c * OUT_DIM + o];
  out[i] = acc;
}

// =====================================================================
static inline int cdiv(long long a, long long b) { return (int)((a + b - 1) / b); }

extern "C" void kernel_launch(void* const* d_in, const int* in_sizes, int n_in,
                              void* d_out, int out_size, void* d_ws, size_t ws_size,
                              hipStream_t stream) {
  (void)n_in; (void)out_size; (void)ws_size;
  const float* x     = (const float*)d_in[0];
  const int*   esrc  = (const int*)  d_in[1];
  const int*   edst  = (const int*)  d_in[2];
  const int*   batch = (const int*)  d_in[3];
  const float* W1    = (const float*)d_in[4];
  const float* asrc1 = (const float*)d_in[5];
  const float* adst1 = (const float*)d_in[6];
  const float* b1    = (const float*)d_in[7];
  const float* W2    = (const float*)d_in[8];
  const float* asrc2 = (const float*)d_in[9];
  const float* adst2 = (const float*)d_in[10];
  const float* b2    = (const float*)d_in[11];
  const float* fc_w  = (const float*)d_in[12];
  const float* fc_b  = (const float*)d_in[13];
  float* out = (float*)d_out;

  const int N  = in_sizes[3];        // batch has one entry per node
  const int E  = in_sizes[1];
  const int Et = E + N;              // with self-loops
  const int F1 = HEADS * HID;        // 128

  // ---- carve workspace (256B aligned slabs) ----
  char* w = (char*)d_ws;
  auto carve = [&](size_t bytes) -> void* {
    void* p = (void*)w; w += (bytes + 255) & ~(size_t)255; return p;
  };
  __bf16* xb   = (__bf16*)carve((size_t)N * F_IN * sizeof(__bf16));
  __bf16* W1T  = (__bf16*)carve((size_t)F1 * F_IN * sizeof(__bf16));
  __bf16* W2T  = (__bf16*)carve((size_t)HID * F1 * sizeof(__bf16));
  float*  h1   = (float*) carve((size_t)N * F1 * sizeof(float));
  float*  as1  = (float*) carve((size_t)N * HEADS * sizeof(float));
  float*  ad1  = (float*) carve((size_t)N * HEADS * sizeof(float));
  float*  m1   = (float*) carve((size_t)N * HEADS * sizeof(float));
  float*  dn1  = (float*) carve((size_t)N * HEADS * sizeof(float));
  float*  lg1  = (float*) carve((size_t)Et * HEADS * sizeof(float));
  float*  o1   = (float*) carve((size_t)N * F1 * sizeof(float));
  __bf16* h1b  = (__bf16*)carve((size_t)N * F1 * sizeof(__bf16));
  float*  h2   = (float*) carve((size_t)N * HID * sizeof(float));
  float*  as2  = (float*) carve((size_t)N * sizeof(float));
  float*  ad2  = (float*) carve((size_t)N * sizeof(float));
  float*  m2   = (float*) carve((size_t)N * sizeof(float));
  float*  dn2  = (float*) carve((size_t)N * sizeof(float));
  float*  lg2  = (float*) carve((size_t)Et * sizeof(float));
  float*  o2   = (float*) carve((size_t)N * HID * sizeof(float));
  float*  pool = (float*) carve((size_t)NGRAPH * HID * sizeof(float));
  float*  cnt  = (float*) carve((size_t)NGRAPH * sizeof(float));

  const int B = 256;
  const float NEG_INF = -__builtin_huge_valf();

  // ---- prep: bf16 casts / weight transposes ----
  cast_bf16_kernel<<<cdiv((long long)N * F_IN, B), B, 0, stream>>>(x, xb, N * F_IN);
  transpose_cast_kernel<<<cdiv(F_IN * F1, B), B, 0, stream>>>(W1, W1T, F_IN, F1);
  transpose_cast_kernel<<<cdiv(F1 * HID, B), B, 0, stream>>>(W2, W2T, F1, HID);

  // ---- layer 1 GEMM: h1 = x @ W1  (WMMA bf16, f32 acc) ----
  {
    long long waves = (long long)cdiv(N, 16) * (F1 / 16);
    wmma_gemm_bf16_kernel<<<cdiv(waves * 32, B), B, 0, stream>>>(xb, W1T, h1, N, F_IN, F1);
  }
  alpha_kernel<<<cdiv((long long)N * HEADS, B), B, 0, stream>>>(h1, asrc1, adst1, as1, ad1, N, HEADS, HID);

  // ---- layer 1 segment softmax + aggregate ----
  fillf_kernel<<<cdiv((long long)N * HEADS, B), B, 0, stream>>>(m1, NEG_INF, N * HEADS);
  fillf_kernel<<<cdiv((long long)N * HEADS, B), B, 0, stream>>>(dn1, 0.f, N * HEADS);
  fillf_kernel<<<cdiv((long long)N * F1, B), B, 0, stream>>>(o1, 0.f, N * F1);
  edge_logit_max_kernel<<<cdiv((long long)Et * HEADS, B), B, 0, stream>>>(esrc, edst, as1, ad1, lg1, m1, E, Et, HEADS);
  edge_expsum_kernel<<<cdiv((long long)Et * HEADS, B), B, 0, stream>>>(edst, lg1, m1, dn1, E, Et, HEADS);
  edge_scatter_kernel<<<cdiv((long long)Et * 32, B), B, 0, stream>>>(esrc, edst, lg1, dn1, h1, o1, E, Et, HEADS, HID);
  relu_bias_cast_kernel<<<cdiv((long long)N * F1, B), B, 0, stream>>>(o1, b1, h1b, N, F1);

  // ---- layer 2 GEMM: h2 = relu(h1) @ W2 (WMMA) ----
  {
    long long waves = (long long)cdiv(N, 16) * (HID / 16);
    wmma_gemm_bf16_kernel<<<cdiv(waves * 32, B), B, 0, stream>>>(h1b, W2T, h2, N, F1, HID);
  }
  alpha_kernel<<<cdiv((long long)N, B), B, 0, stream>>>(h2, asrc2, adst2, as2, ad2, N, 1, HID);

  // ---- layer 2 segment softmax + aggregate ----
  fillf_kernel<<<cdiv((long long)N, B), B, 0, stream>>>(m2, NEG_INF, N);
  fillf_kernel<<<cdiv((long long)N, B), B, 0, stream>>>(dn2, 0.f, N);
  fillf_kernel<<<cdiv((long long)N * HID, B), B, 0, stream>>>(o2, 0.f, N * HID);
  edge_logit_max_kernel<<<cdiv((long long)Et, B), B, 0, stream>>>(esrc, edst, as2, ad2, lg2, m2, E, Et, 1);
  edge_expsum_kernel<<<cdiv((long long)Et, B), B, 0, stream>>>(edst, lg2, m2, dn2, E, Et, 1);
  edge_scatter_kernel<<<cdiv((long long)Et * 32, B), B, 0, stream>>>(esrc, edst, lg2, dn2, h2, o2, E, Et, 1, HID);

  // ---- mean pool + FC head ----
  fillf_kernel<<<cdiv(NGRAPH * HID, B), B, 0, stream>>>(pool, 0.f, NGRAPH * HID);
  fillf_kernel<<<1, NGRAPH, 0, stream>>>(cnt, 0.f, NGRAPH);
  relu_pool_kernel<<<cdiv((long long)N * HID, B), B, 0, stream>>>(o2, b2, batch, pool, cnt, N, HID);
  fc_kernel<<<cdiv(NGRAPH * OUT_DIM, B), B, 0, stream>>>(pool, cnt, fc_w, fc_b, out);
}
